// InEKFLevel15_ConstR_51170240365110
// MI455X (gfx1250) — matrix operand order repaired and checked by reading, not compiled, not run on hardware
//
#include <hip/hip_runtime.h>
#include <math.h>

typedef __attribute__((ext_vector_type(4)))  _Float16 v4h;
typedef __attribute__((ext_vector_type(8)))  _Float16 v8h;
typedef __attribute__((ext_vector_type(16))) _Float16 v16h;
typedef __attribute__((ext_vector_type(8)))  float    v8f;

#define D_MODEL 1024
#define HIDN    128
#define OUTC    256
#define BATCH   4
#define SEQ     4096
#define ROWS    (BATCH * SEQ)      // 16384
#define NELEM   (ROWS * OUTC)      // 4194304
#define CHUNKS  32
#define CLEN    (SEQ / CHUNKS)     // 128
#define KT      128                // K per staged A tile
#define APAD    136                // 128 + 8 halves: 16B-aligned rows, bank stagger

// Build a v16h fragment from two 16-byte loads (LDS or global).
__device__ __forceinline__ v16h frag16(const _Float16* p0, const _Float16* p1) {
  v8h lo = *(const v8h*)p0;
  v8h hi = *(const v8h*)p1;
  return __builtin_shufflevector(lo, hi, 0, 1, 2, 3, 4, 5, 6, 7,
                                 8, 9, 10, 11, 12, 13, 14, 15);
}

// ---------------------------------------------------------------------------
// Kernel 0: one-time weight prep — f16 + transpose so GEMM B-fragments are
// contiguous 16B runs:  W1t[c][k] (128x1024), W2t[c][k] (256x128).
// ---------------------------------------------------------------------------
__global__ __launch_bounds__(256) void prep_wt(const float* __restrict__ W1,
                                               const float* __restrict__ W2,
                                               _Float16* __restrict__ W1t,
                                               _Float16* __restrict__ W2t) {
  int i = blockIdx.x * 256 + threadIdx.x;          // grid covers 131072
  { int k = i & (D_MODEL - 1), c = i >> 10;
    W1t[c * D_MODEL + k] = (_Float16)W1[k * HIDN + c]; }
  if (i < HIDN * OUTC) {
    int k = i & (HIDN - 1), c = i >> 7;
    W2t[c * HIDN + k] = (_Float16)W2[k * OUTC + c];
  }
}

// ---------------------------------------------------------------------------
// Kernel 1: h = GELU(X @ W1 + b1).  Block = 256 thr (8 waves), 16-row strip.
// A tile (16x128) double-buffered in LDS (float4 loads -> v4h stores);
// B fragments straight from pre-transposed f16 W1t in global (L2-resident).
// One barrier per K-step, 4 WMMAs per step.
// ---------------------------------------------------------------------------
__global__ __launch_bounds__(256) void gemm1_gelu(const float* __restrict__ X,
                                                  const _Float16* __restrict__ W1t,
                                                  const float* __restrict__ b1,
                                                  _Float16* __restrict__ Hout) {
  __shared__ alignas(32) _Float16 As[2][16 * APAD];   // ~8.5 KB
  const int tid     = threadIdx.x;
  const int lane    = tid & 31;
  const int rowBase = blockIdx.x * 16;
  const int colBase = (tid >> 5) * 16;

  const int r     = lane & 15;
  const int abase = (lane < 16) ? 0 : 8;
  const int kbase = (lane < 16) ? 0 : 8;
  const int col   = colBase + (lane & 15);

  const int srr = tid >> 5;          // staging row (0..7; +8 for second)
  const int sc4 = (tid & 31) << 2;   // staging col group (4 floats)

  auto stage = [&](int buf, int kk) {
    // batch the loads first (restores memory-level parallelism)
    float4 v0 = *(const float4*)&X[(rowBase + srr)     * D_MODEL + kk + sc4];
    float4 v1 = *(const float4*)&X[(rowBase + 8 + srr) * D_MODEL + kk + sc4];
    v4h h0 = { (_Float16)v0.x, (_Float16)v0.y, (_Float16)v0.z, (_Float16)v0.w };
    v4h h1 = { (_Float16)v1.x, (_Float16)v1.y, (_Float16)v1.z, (_Float16)v1.w };
    *(v4h*)&As[buf][srr       * APAD + sc4] = h0;
    *(v4h*)&As[buf][(8 + srr) * APAD + sc4] = h1;
  };

  stage(0, 0);
  v8f acc = {};
  const int NIT = D_MODEL / KT;      // 8
  for (int it = 0; it < NIT; ++it) {
    const int cur = it & 1;
    __syncthreads();                 // prev compute done + cur tile visible
    if (it + 1 < NIT) stage(cur ^ 1, (it + 1) * KT);
    if (it + 2 < NIT)                // global_prefetch_b8 two tiles ahead
      __builtin_prefetch(&X[(rowBase + srr) * D_MODEL + (it + 2) * KT + sc4], 0, 1);

    const _Float16* wrow = W1t + col * D_MODEL + it * KT;
    #pragma unroll
    for (int ko = 0; ko < KT; ko += 32) {
      v16h a = frag16(&As[cur][r * APAD + ko + abase],
                      &As[cur][r * APAD + ko + abase + 16]);
      v16h b = frag16(wrow + ko + kbase, wrow + ko + kbase + 16);
      acc = __builtin_amdgcn_wmma_f32_16x16x32_f16(false, a, false, b,
                                                   (short)0, acc, false, false);
    }
  }

  // epilogue: bias + exact GELU, store f16
  const int mofs = (lane < 16) ? 0 : 8;
  const float bias = b1[col];
  #pragma unroll
  for (int v = 0; v < 8; ++v) {
    float x = acc[v] + bias;
    float g = 0.5f * x * (1.0f + erff(x * 0.70710678118654752f));
    Hout[(rowBase + v + mofs) * HIDN + col] = (_Float16)g;
  }
}

// ---------------------------------------------------------------------------
// Kernel 2: C = h @ W2 + b2; z = pi*tanh(C); nu = wrap(z - theta); u = K*nu.
// Block = 512 thr (16 waves) covering all 256 cols of a 16-row strip.
// A strip staged once in LDS; B from pre-transposed W2t; single barrier.
// ---------------------------------------------------------------------------
__global__ __launch_bounds__(512) void gemm2_u(const _Float16* __restrict__ Hin,
                                               const _Float16* __restrict__ W2t,
                                               const float* __restrict__ b2,
                                               const float* __restrict__ theta,
                                               const float* __restrict__ logPi,
                                               const float* __restrict__ logR,
                                               float* __restrict__ out) {
  __shared__ alignas(32) _Float16 As[16 * APAD];   // ~4.3 KB
  const int tid     = threadIdx.x;
  const int lane    = tid & 31;
  const int rowBase = blockIdx.x * 16;
  const int colBase = (tid >> 5) * 16;

  const int r     = lane & 15;
  const int abase = (lane < 16) ? 0 : 8;
  const int kbase = (lane < 16) ? 0 : 8;
  const int col   = colBase + (lane & 15);

  if (tid < 256) {                   // stage 16x128 f16 strip as v8h copies
    int g = tid & 15, rr = tid >> 4;
    *(v8h*)&As[rr * APAD + g * 8] =
        *(const v8h*)&Hin[(rowBase + rr) * HIDN + g * 8];
  }
  __syncthreads();

  v8f acc = {};
  const _Float16* wrow = W2t + col * HIDN;
  #pragma unroll
  for (int ko = 0; ko < HIDN; ko += 32) {
    v16h a = frag16(&As[r * APAD + ko + abase],
                    &As[r * APAD + ko + abase + 16]);
    v16h b = frag16(wrow + ko + kbase, wrow + ko + kbase + 16);
    acc = __builtin_amdgcn_wmma_f32_16x16x32_f16(false, a, false, b,
                                                 (short)0, acc, false, false);
  }

  const int mofs = (lane < 16) ? 0 : 8;
  const float bias = b2[col];
  const float Pi = expf(logPi[col]);
  const float Rv = expf(logR[col]);
  const float Kv = Pi / fmaxf(Pi + Rv, 1e-8f);
  float* out0 = out;
  float* out1 = out + (size_t)NELEM;
  float* out2 = out + (size_t)2 * NELEM;
  float* out3 = out + (size_t)3 * NELEM;
  #pragma unroll
  for (int v = 0; v < 8; ++v) {
    int row = rowBase + v + mofs;
    int idx = row * OUTC + col;
    float z    = 3.14159265358979323846f * tanhf(acc[v] + bias);
    float diff = z - theta[idx];
    float nu   = atan2f(sinf(diff), cosf(diff));
    out0[idx] = Kv * nu;   // u (scratch; rewritten by scan_apply)
    out1[idx] = Pi;
    out2[idx] = Kv;
    out3[idx] = Rv;
  }
}

// ---------------------------------------------------------------------------
// Blocked scan: d_t = a*d_{t-1} + u_t, a = 1-K constant per channel.
// ---------------------------------------------------------------------------
__device__ __forceinline__ float channel_alpha(const float* logPi,
                                               const float* logR, int col) {
  float Pi = expf(logPi[col]);
  float Rv = expf(logR[col]);
  return 1.0f - Pi / fmaxf(Pi + Rv, 1e-8f);
}

__global__ __launch_bounds__(256) void scan_chunks(const float* __restrict__ u,
                                                   const float* __restrict__ logPi,
                                                   const float* __restrict__ logR,
                                                   float* __restrict__ states) {
  const int col   = threadIdx.x;
  const int chunk = blockIdx.x % CHUNKS;
  const int b     = blockIdx.x / CHUNKS;
  const float a = channel_alpha(logPi, logR, col);
  float s = 0.0f;
  int base = (b * SEQ + chunk * CLEN) * OUTC + col;
  for (int l = 0; l < CLEN; ++l)
    s = a * s + u[base + l * OUTC];
  states[(b * CHUNKS + chunk) * OUTC + col] = s;
}

__global__ __launch_bounds__(256) void scan_carries(const float* __restrict__ states,
                                                    const float* __restrict__ logPi,
                                                    const float* __restrict__ logR,
                                                    float* __restrict__ carries) {
  const int col = threadIdx.x;
  const int b   = blockIdx.x;
  const float a = channel_alpha(logPi, logR, col);
  float aP = a;
  #pragma unroll
  for (int i = 0; i < 7; ++i) aP *= aP;   // a^128
  float carry = 0.0f;
  for (int c = 0; c < CHUNKS; ++c) {
    int idx = (b * CHUNKS + c) * OUTC + col;
    carries[idx] = carry;
    carry = aP * carry + states[idx];
  }
}

__global__ __launch_bounds__(256) void scan_apply(const float* __restrict__ theta,
                                                  const float* __restrict__ logPi,
                                                  const float* __restrict__ logR,
                                                  const float* __restrict__ carries,
                                                  float* __restrict__ out0) {
  const int col   = threadIdx.x;
  const int chunk = blockIdx.x % CHUNKS;
  const int b     = blockIdx.x / CHUNKS;
  const float a = channel_alpha(logPi, logR, col);
  float d = carries[(b * CHUNKS + chunk) * OUTC + col];
  int base = (b * SEQ + chunk * CLEN) * OUTC + col;
  for (int l = 0; l < CLEN; ++l) {
    int idx = base + l * OUTC;
    d = a * d + out0[idx];             // read u
    out0[idx] = theta[idx] + d;        // overwrite with final output
  }
}

// ---------------------------------------------------------------------------
extern "C" void kernel_launch(void* const* d_in, const int* in_sizes, int n_in,
                              void* d_out, int out_size, void* d_ws, size_t ws_size,
                              hipStream_t stream) {
  (void)in_sizes; (void)n_in; (void)out_size; (void)ws_size;
  const float* theta = (const float*)d_in[0];
  const float* X     = (const float*)d_in[1];
  const float* W1    = (const float*)d_in[2];
  const float* b1    = (const float*)d_in[3];
  const float* W2    = (const float*)d_in[4];
  const float* b2    = (const float*)d_in[5];
  const float* logPi = (const float*)d_in[6];
  const float* logR  = (const float*)d_in[7];
  float* out = (float*)d_out;

  char* ws = (char*)d_ws;
  _Float16* h_ws  = (_Float16*)ws;                                   // 4 MB
  _Float16* W1t   = (_Float16*)(ws + (size_t)ROWS * HIDN * 2);       // 256 KB
  _Float16* W2t   = W1t + (size_t)D_MODEL * HIDN;                    // 64 KB
  float* states   = (float*)(W2t + (size_t)HIDN * OUTC);             // 128 KB
  float* carries  = states + BATCH * CHUNKS * OUTC;                  // 128 KB

  prep_wt     <<<(D_MODEL * HIDN) / 256, 256, 0, stream>>>(W1, W2, W1t, W2t);
  gemm1_gelu  <<<ROWS / 16, 256, 0, stream>>>(X, W1t, b1, h_ws);
  gemm2_u     <<<ROWS / 16, 512, 0, stream>>>(h_ws, W2t, b2, theta, logPi, logR, out);
  scan_chunks <<<BATCH * CHUNKS, 256, 0, stream>>>(out, logPi, logR, states);
  scan_carries<<<BATCH, 256, 0, stream>>>(states, logPi, logR, carries);
  scan_apply  <<<BATCH * CHUNKS, 256, 0, stream>>>(theta, logPi, logR, carries, out);
}
